// VarianceAdaptor_36524401886014
// MI455X (gfx1250) — matrix operand order repaired and checked by reading, not compiled, not run on hardware
//
#include <hip/hip_runtime.h>
#include <hip/hip_bf16.h>

// ---------------- problem constants (match reference) ----------------
#define BB      32
#define TT      512
#define DD      512
#define KK      3
#define NBINS   256
#define MAXLEN  4096
#define TP      (TT + 2)        // padded rows for K=3 "same" conv
#define CDIM    (KK * DD)       // 1536 = GEMM reduction dim
#define LN_EPS  1e-5f

typedef __attribute__((ext_vector_type(16))) __bf16 v16bf;
typedef __attribute__((ext_vector_type(8)))  __bf16 v8bf;
typedef __attribute__((ext_vector_type(8)))  float  v8f;

// ---------------- helpers ----------------
__device__ __forceinline__ unsigned short f2bf(float f) {
    unsigned int u = __float_as_uint(f);
    unsigned int r = u + 0x7FFFu + ((u >> 16) & 1u);   // round-to-nearest-even
    return (unsigned short)(r >> 16);
}

__device__ __forceinline__ v16bf cat8(v8bf lo, v8bf hi) {
    return __builtin_shufflevector(lo, hi, 0,1,2,3,4,5,6,7,8,9,10,11,12,13,14,15);
}

// ---------------- 1) convert x -> padded bf16; zero pad rows of both padded bufs ----
__global__ void cvt_x_kernel(const float* __restrict__ x,
                             unsigned short* __restrict__ xpad,
                             unsigned short* __restrict__ h2pad) {
    size_t idx = (size_t)blockIdx.x * blockDim.x + threadIdx.x;
    size_t total = (size_t)BB * TP * DD;
    if (idx >= total) return;
    int d = (int)(idx % DD);
    int r = (int)((idx / DD) % TP);
    int b = (int)(idx / ((size_t)DD * TP));
    if (r == 0 || r == TP - 1) {
        xpad[idx]  = 0;
        h2pad[idx] = 0;
    } else {
        xpad[idx] = f2bf(x[((size_t)b * TT + (r - 1)) * DD + d]);
    }
}

// ---------------- 2) convert conv weight (D,D,K) f32 -> Wt[dout][k*D+din] bf16 ----
__global__ void cvt_w_kernel(const float* __restrict__ w, unsigned short* __restrict__ wt) {
    size_t idx = (size_t)blockIdx.x * blockDim.x + threadIdx.x;
    size_t total = (size_t)DD * CDIM;
    if (idx >= total) return;
    int c    = (int)(idx % CDIM);
    int dout = (int)(idx / CDIM);
    int k   = c >> 9;         // c / DD
    int din = c & (DD - 1);   // c % DD
    wt[idx] = f2bf(w[((size_t)dout * DD + din) * KK + k]);
}

// ---------------- 3) conv1d as WMMA GEMM: out = relu(A[16384x1536]*Wt^T + bias) ----
// grid: (B*T/64, 2), block: 128 (4 waves).
// Each wave owns a 64x64 C tile: 4 M-tiles x 4 N-tiles = 16 f32 accumulators.
// Per K-step: 8 A b128 loads + 8 B b128 loads -> 16 v_wmma (1:1 load:wmma).
__global__ __launch_bounds__(128) void conv_wmma_kernel(
        const __bf16* __restrict__ Apad,   // [B][T+2][D] bf16, rows 0 / T+1 are zero
        const __bf16* __restrict__ Wt,     // [D][3*D] bf16
        const float*  __restrict__ bias,   // [D]
        float*        __restrict__ out)    // [B][T][D] f32 (relu applied)
{
    const int lane  = threadIdx.x & 31;
    const int wave  = threadIdx.x >> 5;
    const int mg    = blockIdx.x;              // M-group: 64 rows
    const int b     = mg >> 3;                 // 512/64 == 8 M-groups per batch row
    const int t0    = (mg & 7) << 6;
    const int g     = blockIdx.y * 4 + wave;   // n-group 0..7 -> douts [g*64, g*64+64)
    const int n     = lane & 15;
    const int m     = lane & 15;
    const int khalf = lane >> 4;

    v8f acc[4][4];
#pragma unroll
    for (int mi = 0; mi < 4; ++mi)
#pragma unroll
        for (int ni = 0; ni < 4; ++ni)
            acc[mi][ni] = v8f{};

    const __bf16* arowBase = Apad + (size_t)b * TP * DD;
    const __bf16* wbase    = Wt + (size_t)((g << 6) + n) * CDIM + khalf * 16;

    for (int kk = 0; kk < CDIM / 32; ++kk) {           // 48 K-steps of 32
        const int c0   = kk << 5;
        const int k    = c0 >> 9;
        const int din0 = c0 & (DD - 1);

        // prefetch next weight chunk into WGP$ (global_prefetch_b8)
        __builtin_prefetch(wbase + ((c0 + 32) % CDIM), 0, 0);

        // A tiles 16x32 bf16: per-lane 16 contiguous K (k = 16*khalf + s), matching B
        v16bf a[4];
#pragma unroll
        for (int mi = 0; mi < 4; ++mi) {
            const __bf16* arow =
                arowBase + (size_t)(t0 + (mi << 4) + m + k) * DD + din0;
            v8bf alo = *(const v8bf*)(arow + khalf * 16);
            v8bf ahi = *(const v8bf*)(arow + khalf * 16 + 8);
            a[mi] = cat8(alo, ahi);
        }

#pragma unroll
        for (int ni = 0; ni < 4; ++ni) {
            // B tile 32x16 bf16: lane reads 16 contiguous K of column (dout0+n)
            const __bf16* wrow = wbase + (size_t)(ni << 4) * CDIM + c0;
            v8bf blo = *(const v8bf*)(wrow);
            v8bf bhi = *(const v8bf*)(wrow + 8);
            v16bf bm = cat8(blo, bhi);
#pragma unroll
            for (int mi = 0; mi < 4; ++mi) {
                acc[mi][ni] = __builtin_amdgcn_wmma_f32_16x16x32_bf16(
                                  false, a[mi], false, bm, (short)0,
                                  acc[mi][ni], false, false);
            }
        }
    }

    // epilogue: C layout lane->(m = v + 8*khalf, n = lane&15); bias + relu
#pragma unroll
    for (int ni = 0; ni < 4; ++ni) {
        const int dout0 = (g << 6) + (ni << 4);
        const float bv = bias[dout0 + n];
#pragma unroll
        for (int mi = 0; mi < 4; ++mi) {
#pragma unroll
            for (int v = 0; v < 8; ++v) {
                const int mm = (mi << 4) + v + (khalf << 3);
                float val = acc[mi][ni][v] + bv;
                val = fmaxf(val, 0.0f);
                out[((size_t)b * TT + t0 + mm) * DD + dout0 + n] = val;
            }
        }
    }
}

// ---------------- 4) layernorm (f32 in) -> bf16 padded out (feeds next conv) ----
__global__ __launch_bounds__(256) void ln_kernel(const float* __restrict__ h,
                                                 const float* __restrict__ g,
                                                 const float* __restrict__ be,
                                                 unsigned short* __restrict__ outpad) {
    const int row = blockIdx.x;                 // 0..B*T-1
    const int b = row / TT, t = row % TT;
    const int tid = threadIdx.x;
    __shared__ float ssum[256], ssq[256];
    const float* hr = h + (size_t)row * DD;
    float a0 = hr[tid], a1 = hr[tid + 256];
    ssum[tid] = a0 + a1;
    ssq[tid]  = a0 * a0 + a1 * a1;
    __syncthreads();
    for (int off = 128; off > 0; off >>= 1) {
        if (tid < off) { ssum[tid] += ssum[tid + off]; ssq[tid] += ssq[tid + off]; }
        __syncthreads();
    }
    __shared__ float mstat[2];
    if (tid == 0) {
        float mean = ssum[0] * (1.0f / DD);
        float var  = ssq[0] * (1.0f / DD) - mean * mean;
        mstat[0] = mean;
        mstat[1] = rsqrtf(var + LN_EPS);
    }
    __syncthreads();
    const float mean = mstat[0], rstd = mstat[1];
    unsigned short* orow = outpad + ((size_t)b * TP + t + 1) * DD;
    orow[tid]       = f2bf((a0 - mean) * rstd * g[tid]       + be[tid]);
    orow[tid + 256] = f2bf((a1 - mean) * rstd * g[tid + 256] + be[tid + 256]);
}

// ---------------- 5) layernorm + linear(D->1) + mask -> pred[b,t] ----
__global__ __launch_bounds__(256) void ln_linear_kernel(const float* __restrict__ h,
                                                        const float* __restrict__ g,
                                                        const float* __restrict__ be,
                                                        const float* __restrict__ lw,
                                                        const float* __restrict__ lb,
                                                        const unsigned char* __restrict__ mask,
                                                        float* __restrict__ pred) {
    const int row = blockIdx.x;
    const int tid = threadIdx.x;
    __shared__ float ssum[256], ssq[256];
    const float* hr = h + (size_t)row * DD;
    float a0 = hr[tid], a1 = hr[tid + 256];
    ssum[tid] = a0 + a1;
    ssq[tid]  = a0 * a0 + a1 * a1;
    __syncthreads();
    for (int off = 128; off > 0; off >>= 1) {
        if (tid < off) { ssum[tid] += ssum[tid + off]; ssq[tid] += ssq[tid + off]; }
        __syncthreads();
    }
    __shared__ float mstat[2];
    if (tid == 0) {
        float mean = ssum[0] * (1.0f / DD);
        float var  = ssq[0] * (1.0f / DD) - mean * mean;
        mstat[0] = mean;
        mstat[1] = rsqrtf(var + LN_EPS);
    }
    __syncthreads();
    const float mean = mstat[0], rstd = mstat[1];
    float c0 = ((a0 - mean) * rstd * g[tid]       + be[tid])       * lw[tid];
    float c1 = ((a1 - mean) * rstd * g[tid + 256] + be[tid + 256]) * lw[tid + 256];
    __syncthreads();
    ssum[tid] = c0 + c1;
    __syncthreads();
    for (int off = 128; off > 0; off >>= 1) {
        if (tid < off) ssum[tid] += ssum[tid + off];
        __syncthreads();
    }
    if (tid == 0)
        pred[row] = mask[row] ? (ssum[0] + lb[0]) : 0.0f;
}

// ---------------- 6) h = x + mask*(pitch_emb[bin] + energy_emb[bin]) ----
__global__ __launch_bounds__(256) void embed_add_kernel(const float* __restrict__ x,
                                                        const float* __restrict__ pitch,
                                                        const float* __restrict__ energy,
                                                        const float* __restrict__ pemb,
                                                        const float* __restrict__ eemb,
                                                        const unsigned char* __restrict__ mask,
                                                        float* __restrict__ h) {
    const int row = blockIdx.x;                  // 0..B*T-1
    const int tid = threadIdx.x;
    float pv = fminf(fmaxf(pitch[row], 0.0f), 800.0f);
    float ev = fminf(fmaxf(energy[row], 0.0f), 100.0f);
    int pb = (int)(pv * ((NBINS - 1) / 800.0f));
    int eb = (int)(ev * ((NBINS - 1) / 100.0f));
    pb = min(max(pb, 0), NBINS - 1);
    eb = min(max(eb, 0), NBINS - 1);
    const float mk = mask[row] ? 1.0f : 0.0f;
    const float* xr = x + (size_t)row * DD;
    const float* pr = pemb + (size_t)pb * DD;
    const float* er = eemb + (size_t)eb * DD;
    float* hr = h + (size_t)row * DD;
    hr[tid]       = xr[tid]       + mk * (pr[tid]       + er[tid]);
    hr[tid + 256] = xr[tid + 256] + mk * (pr[tid + 256] + er[tid + 256]);
}

// ---------------- 7) per-batch inclusive cumsum of duration; mel_lengths ----
__global__ __launch_bounds__(512) void cumsum_kernel(const int* __restrict__ duration,
                                                     int* __restrict__ cum,
                                                     int* __restrict__ totals,
                                                     float* __restrict__ mel_len_out) {
    const int b = blockIdx.x;
    const int tid = threadIdx.x;
    __shared__ int s[TT];
    s[tid] = duration[(size_t)b * TT + tid];
    __syncthreads();
    for (int off = 1; off < TT; off <<= 1) {
        int v = (tid >= off) ? s[tid - off] : 0;
        __syncthreads();
        s[tid] += v;
        __syncthreads();
    }
    cum[(size_t)b * TT + tid] = s[tid];
    if (tid == 0) {
        int total = s[TT - 1];
        totals[b] = total;
        mel_len_out[b] = (float)max(total, 1);
    }
}

// ---------------- 8) length-regulate gather + mel_mask ----
__global__ __launch_bounds__(256) void gather_kernel(const float* __restrict__ h,
                                                     const int* __restrict__ cum,
                                                     const int* __restrict__ totals,
                                                     float* __restrict__ out,
                                                     float* __restrict__ mel_mask) {
    const int blk = blockIdx.x;                  // 0..B*MAXLEN-1
    const int b = blk / MAXLEN, tm = blk % MAXLEN;
    const int tid = threadIdx.x;
    __shared__ int s_idx, s_valid;
    if (tid == 0) {
        const int* c = cum + (size_t)b * TT;
        int lo = 0, hi = TT;
        while (lo < hi) {                        // searchsorted(c, tm, 'right')
            int mid = (lo + hi) >> 1;
            if (c[mid] <= tm) lo = mid + 1; else hi = mid;
        }
        s_idx = min(lo, TT - 1);
        int total = totals[b];
        s_valid = (tm < total) ? 1 : 0;
        mel_mask[(size_t)b * MAXLEN + tm] = (tm < max(total, 1)) ? 1.0f : 0.0f;
    }
    __syncthreads();
    const int idx = s_idx, valid = s_valid;
    const float* src = h + ((size_t)b * TT + idx) * DD;
    float* dst = out + ((size_t)b * MAXLEN + tm) * DD;
    dst[tid]       = valid ? src[tid]       : 0.0f;
    dst[tid + 256] = valid ? src[tid + 256] : 0.0f;
}

// ---------------- host launcher ----------------
extern "C" void kernel_launch(void* const* d_in, const int* in_sizes, int n_in,
                              void* d_out, int out_size, void* d_ws, size_t ws_size,
                              hipStream_t stream) {
    (void)in_sizes; (void)n_in; (void)out_size; (void)ws_size;

    // inputs (setup_inputs dict order; tuples flattened)
    const float* x        = (const float*)d_in[0];
    const unsigned char* tmask = (const unsigned char*)d_in[1];   // jnp.bool_
    const int*   duration = (const int*)d_in[2];
    const float* pitch    = (const float*)d_in[3];
    const float* energy   = (const float*)d_in[4];
    const float* pemb     = (const float*)d_in[6];
    const float* eemb     = (const float*)d_in[7];
    // predictor params: 10 arrays each, dp @8, pp @18, ep @28
    const int PBASE[3] = {8, 18, 28};

    // output regions (return order, flattened f32)
    float* out_main = (float*)d_out;
    const size_t OFF_MASK = (size_t)BB * MAXLEN * DD;
    const size_t OFF_DUR  = OFF_MASK + (size_t)BB * MAXLEN;
    const size_t OFF_PIT  = OFF_DUR  + (size_t)BB * TT;
    const size_t OFF_ENE  = OFF_PIT  + (size_t)BB * TT;
    const size_t OFF_MLEN = OFF_ENE  + (size_t)BB * TT;
    float* out_mask = out_main + OFF_MASK;
    float* out_pred[3] = {out_main + OFF_DUR, out_main + OFF_PIT, out_main + OFF_ENE};
    float* out_mlen = out_main + OFF_MLEN;

    // workspace layout
    char* ws = (char*)d_ws;
    const size_t XP = (size_t)BB * TP * DD * 2;        // padded bf16 activations
    const size_t HB = (size_t)BB * TT * DD * 4;        // f32 hidden buffer
    const size_t WT = (size_t)DD * CDIM * 2;           // one bf16 transposed weight
    unsigned short* xpad  = (unsigned short*)(ws);
    unsigned short* h2pad = (unsigned short*)(ws + XP);
    float*          hbuf  = (float*)(ws + 2 * XP);
    unsigned short* wts   = (unsigned short*)(ws + 2 * XP + HB);   // 6 slots
    int*            cum    = (int*)(ws + 2 * XP + HB + 6 * WT);
    int*            totals = (int*)(ws + 2 * XP + HB + 6 * WT + (size_t)BB * TT * 4);

    // 1) x -> padded bf16 (and zero the pad rows of both padded buffers)
    {
        size_t total = (size_t)BB * TP * DD;
        cvt_x_kernel<<<(unsigned)((total + 255) / 256), 256, 0, stream>>>(x, xpad, h2pad);
    }
    // 2) weight conversions: 6 conv weights (dp/pp/ep c1w, c2w)
    for (int p = 0; p < 3; ++p) {
        const float* c1w = (const float*)d_in[PBASE[p] + 0];
        const float* c2w = (const float*)d_in[PBASE[p] + 4];
        size_t total = (size_t)DD * CDIM;
        cvt_w_kernel<<<(unsigned)((total + 255) / 256), 256, 0, stream>>>(
            c1w, wts + (size_t)(2 * p + 0) * (WT / 2));
        cvt_w_kernel<<<(unsigned)((total + 255) / 256), 256, 0, stream>>>(
            c2w, wts + (size_t)(2 * p + 1) * (WT / 2));
    }

    // 3) three variance predictors
    const dim3 cgrid(BB * TT / 64, 2), cblk(128);
    for (int p = 0; p < 3; ++p) {
        const float* c1b = (const float*)d_in[PBASE[p] + 1];
        const float* n1w = (const float*)d_in[PBASE[p] + 2];
        const float* n1b = (const float*)d_in[PBASE[p] + 3];
        const float* c2b = (const float*)d_in[PBASE[p] + 5];
        const float* n2w = (const float*)d_in[PBASE[p] + 6];
        const float* n2b = (const float*)d_in[PBASE[p] + 7];
        const float* lw  = (const float*)d_in[PBASE[p] + 8];
        const float* lb  = (const float*)d_in[PBASE[p] + 9];
        const __bf16* w1 = (const __bf16*)(wts + (size_t)(2 * p + 0) * (WT / 2));
        const __bf16* w2 = (const __bf16*)(wts + (size_t)(2 * p + 1) * (WT / 2));

        conv_wmma_kernel<<<cgrid, cblk, 0, stream>>>((const __bf16*)xpad, w1, c1b, hbuf);
        ln_kernel<<<BB * TT, 256, 0, stream>>>(hbuf, n1w, n1b, h2pad);
        conv_wmma_kernel<<<cgrid, cblk, 0, stream>>>((const __bf16*)h2pad, w2, c2b, hbuf);
        ln_linear_kernel<<<BB * TT, 256, 0, stream>>>(hbuf, n2w, n2b, lw, lb, tmask, out_pred[p]);
    }

    // 4) h = x + masked embeddings (reuses hbuf)
    embed_add_kernel<<<BB * TT, 256, 0, stream>>>(x, pitch, energy, pemb, eemb, tmask, hbuf);

    // 5) cumsum durations -> cum, totals, mel_lengths
    cumsum_kernel<<<BB, TT, 0, stream>>>(duration, cum, totals, out_mlen);

    // 6) length-regulate gather + mel_mask
    gather_kernel<<<BB * MAXLEN, 256, 0, stream>>>(hbuf, cum, totals, out_main, out_mask);
}
